// RnnSimpleRepresentationModel_74062416053088
// MI455X (gfx1250) — matrix operand order repaired and checked by reading, not compiled, not run on hardware
//
#include <hip/hip_runtime.h>
#include <hip/hip_bf16.h>

// Fused 2-layer RNN + FC head for MI455X (gfx1250, wave32, WMMA).
//
// Round-2 design: one WAVE per 16-row batch tile, zero barriers.
//  - grid = 64 blocks x 32 threads (1 wave); wave computes the FULL 16x64
//    hidden state of both layers each timestep (28 v_wmma per step, 4-way
//    independent N-tile chains for ILP).
//  - hidden-state C->A layout conversion goes through per-block LDS with
//    same-wave DS ordering (DScnt) only: no s_barrier in the whole kernel.
//  - all weights resident in VGPRs as f16 WMMA B-tiles (f32 -> f16 once).
//  - x streamed straight from HBM into A-layout registers (full 128B-line
//    utilization per wave) + global_prefetch of the next timestep.

#define TT   512
#define BB   1024
#define INW  32
#define HH   64
#define OUTW 32
#define HP   80   // LDS row stride in halfs: 160 B (16B-aligned rows, spreads banks)

typedef __attribute__((ext_vector_type(16))) _Float16 v16h;
typedef __attribute__((ext_vector_type(8)))  _Float16 v8h;
typedef __attribute__((ext_vector_type(8)))  float    v8f;

#if __has_builtin(__builtin_amdgcn_tanhf)
__device__ __forceinline__ float fast_tanh(float x) { return __builtin_amdgcn_tanhf(x); }
#else
__device__ __forceinline__ float fast_tanh(float x) { return tanhf(x); }
#endif

__device__ __forceinline__ v8f wmma16(v16h a, v16h b, v8f c) {
  // D = A(16x32 f16) * B(32x16 f16) + C(16x16 f32)
  return __builtin_amdgcn_wmma_f32_16x16x32_f16(false, a, false, b, (short)0, c,
                                                false, false);
}

// B-tile (32x16, KxN) in f16 from row-major f32 weight W[N][Kfull] for act @ W^T:
// B[k][n] = W[n0+n][k0+k].  Wave32 layout: lane%16 selects N; lanes 0-15 hold
// K{0..7,16..23}, lanes 16-31 hold K{8..15,24..31} (matches A-tile order).
__device__ __forceinline__ v16h load_Btile(const float* __restrict__ W, int Kfull,
                                           int n0, int k0, int rl, int hi) {
  const float* p = W + (size_t)(n0 + rl) * Kfull + (k0 + hi * 8);
  v16h b;
#pragma unroll
  for (int j = 0; j < 16; ++j) b[j] = (_Float16)p[j < 8 ? j : j + 8];
  return b;
}

// A-tile (16x32, MxK) f16 from a row-major LDS buffer (stride HP halfs).
// Layout: lane%16 = M row; lane<16 holds K{0..7,16..23}, lane>=16 holds
// K{8..15,24..31} -> two aligned 16B DS loads per lane.
__device__ __forceinline__ v16h load_Atile_lds(const _Float16* hbuf, int k0,
                                               int rl, int hi) {
  const _Float16* p = hbuf + rl * HP + k0 + hi * 8;
  v8h lo = *(const v8h*)(p);
  v8h hh = *(const v8h*)(p + 16);
  return __builtin_shufflevector(lo, hh, 0, 1, 2, 3, 4, 5, 6, 7,
                                 8, 9, 10, 11, 12, 13, 14, 15);
}

__global__ __launch_bounds__(32) void rnn_fused_kernel(
    const float* __restrict__ x,
    const float* __restrict__ Wih1, const float* __restrict__ Whh1,
    const float* __restrict__ bih1, const float* __restrict__ bhh1,
    const float* __restrict__ Wih2, const float* __restrict__ Whh2,
    const float* __restrict__ bih2, const float* __restrict__ bhh2,
    const float* __restrict__ Wfc1, const float* __restrict__ bfc1,
    const float* __restrict__ Wfc2, const float* __restrict__ bfc2,
    float* __restrict__ out) {
  // Per-block (== per-wave) staging buffers; same-wave DS ordering means no
  // barriers and no ping-pong are needed.
  __shared__ __align__(16) _Float16 h1buf[16 * HP];
  __shared__ __align__(16) _Float16 h2buf[16 * HP];

  const int lane = threadIdx.x & 31;
  const int rl   = lane & 15;
  const int hi   = lane >> 4;
  const int b0   = blockIdx.x * 16;

  // ---- resident f16 B-layout weights: 28 tiles, loaded once ----
  v16h Bih1[4], Bhh10[4], Bhh11[4], Bih20[4], Bih21[4], Bhh20[4], Bhh21[4];
  float b1[4], b2[4];
#pragma unroll
  for (int n = 0; n < 4; ++n) {
    Bih1[n]  = load_Btile(Wih1, INW, 16 * n, 0,  rl, hi);
    Bhh10[n] = load_Btile(Whh1, HH,  16 * n, 0,  rl, hi);
    Bhh11[n] = load_Btile(Whh1, HH,  16 * n, 32, rl, hi);
    Bih20[n] = load_Btile(Wih2, HH,  16 * n, 0,  rl, hi);
    Bih21[n] = load_Btile(Wih2, HH,  16 * n, 32, rl, hi);
    Bhh20[n] = load_Btile(Whh2, HH,  16 * n, 0,  rl, hi);
    Bhh21[n] = load_Btile(Whh2, HH,  16 * n, 32, rl, hi);
    b1[n] = bih1[16 * n + rl] + bhh1[16 * n + rl];  // fused biases (C: N = lane%16)
    b2[n] = bih2[16 * n + rl] + bhh2[16 * n + rl];
  }

  v16h h1A0 = {}, h1A1 = {};  // A-layout hidden states, h0 = 0
  v16h h2A0 = {}, h2A1 = {};

  const float* xrow = x + (size_t)(b0 + rl) * TT * INW;  // this lane's batch row

  for (int t = 0; t < TT; ++t) {
    // ------------- layer 1: h1 = tanh(x_t Wih1^T + b + h1 Whh1^T) -----------
    const float* xp = xrow + t * INW + hi * 8;
    if (t + 1 < TT) __builtin_prefetch(xp + INW, 0, 1);  // global_prefetch
    v16h xa;
#pragma unroll
    for (int j = 0; j < 8; ++j) {            // A-layout K-chunks {hi*8, 16+hi*8}
      xa[j]     = (_Float16)xp[j];
      xa[j + 8] = (_Float16)xp[16 + j];
    }
    v8f a1[4];
#pragma unroll
    for (int n = 0; n < 4; ++n) {            // 4 independent accumulator chains
#pragma unroll
      for (int r = 0; r < 8; ++r) a1[n][r] = b1[n];
      a1[n] = wmma16(xa,   Bih1[n],  a1[n]);
      a1[n] = wmma16(h1A0, Bhh10[n], a1[n]);
      a1[n] = wmma16(h1A1, Bhh11[n], a1[n]);
    }
#pragma unroll
    for (int n = 0; n < 4; ++n)              // C-layout: row = r+8*hi, col = 16n+rl
#pragma unroll
      for (int r = 0; r < 8; ++r)
        h1buf[(r + 8 * hi) * HP + 16 * n + rl] = (_Float16)fast_tanh(a1[n][r]);
    // same-wave DS ordering: stores complete before these loads return
    h1A0 = load_Atile_lds(h1buf, 0,  rl, hi);
    h1A1 = load_Atile_lds(h1buf, 32, rl, hi);

    // ------------- layer 2: h2 = tanh(h1 Wih2^T + b + h2 Whh2^T) ------------
    v8f a2[4];
#pragma unroll
    for (int n = 0; n < 4; ++n) {
#pragma unroll
      for (int r = 0; r < 8; ++r) a2[n][r] = b2[n];
      a2[n] = wmma16(h1A0, Bih20[n], a2[n]);
      a2[n] = wmma16(h1A1, Bih21[n], a2[n]);
      a2[n] = wmma16(h2A0, Bhh20[n], a2[n]);
      a2[n] = wmma16(h2A1, Bhh21[n], a2[n]);
    }
#pragma unroll
    for (int n = 0; n < 4; ++n)
#pragma unroll
      for (int r = 0; r < 8; ++r)
        h2buf[(r + 8 * hi) * HP + 16 * n + rl] = (_Float16)fast_tanh(a2[n][r]);
    h2A0 = load_Atile_lds(h2buf, 0,  rl, hi);
    h2A1 = load_Atile_lds(h2buf, 32, rl, hi);
  }

  // ------------- FC head: out = relu(h2 Wfc1^T + b) Wfc2^T + b --------------
  // h2A0/h2A1 already hold the final hidden state in A-layout.
#pragma unroll
  for (int n = 0; n < 4; ++n) {
    v16h Bf0 = load_Btile(Wfc1, HH, 16 * n, 0,  rl, hi);
    v16h Bf1 = load_Btile(Wfc1, HH, 16 * n, 32, rl, hi);
    v8f az;
#pragma unroll
    for (int r = 0; r < 8; ++r) az[r] = bfc1[16 * n + rl];
    az = wmma16(h2A0, Bf0, az);
    az = wmma16(h2A1, Bf1, az);
#pragma unroll
    for (int r = 0; r < 8; ++r)              // z in f16, staged in h1buf
      h1buf[(r + 8 * hi) * HP + 16 * n + rl] = (_Float16)fmaxf(az[r], 0.0f);
  }
  v16h zA0 = load_Atile_lds(h1buf, 0,  rl, hi);
  v16h zA1 = load_Atile_lds(h1buf, 32, rl, hi);
#pragma unroll
  for (int n = 0; n < 2; ++n) {              // OUT = 32 -> 2 N-tiles
    v16h Bf0 = load_Btile(Wfc2, HH, 16 * n, 0,  rl, hi);
    v16h Bf1 = load_Btile(Wfc2, HH, 16 * n, 32, rl, hi);
    v8f ao;
#pragma unroll
    for (int r = 0; r < 8; ++r) ao[r] = bfc2[16 * n + rl];
    ao = wmma16(zA0, Bf0, ao);
    ao = wmma16(zA1, Bf1, ao);
#pragma unroll
    for (int r = 0; r < 8; ++r)
      out[(size_t)(b0 + r + 8 * hi) * OUTW + 16 * n + rl] = ao[r];
  }
}

extern "C" void kernel_launch(void* const* d_in, const int* in_sizes, int n_in,
                              void* d_out, int out_size, void* d_ws, size_t ws_size,
                              hipStream_t stream) {
  (void)in_sizes; (void)n_in; (void)out_size; (void)d_ws; (void)ws_size;
  const float* x    = (const float*)d_in[0];
  const float* Wih1 = (const float*)d_in[1];
  const float* Whh1 = (const float*)d_in[2];
  const float* bih1 = (const float*)d_in[3];
  const float* bhh1 = (const float*)d_in[4];
  const float* Wih2 = (const float*)d_in[5];
  const float* Whh2 = (const float*)d_in[6];
  const float* bih2 = (const float*)d_in[7];
  const float* bhh2 = (const float*)d_in[8];
  const float* Wfc1 = (const float*)d_in[9];
  const float* bfc1 = (const float*)d_in[10];
  const float* Wfc2 = (const float*)d_in[11];
  const float* bfc2 = (const float*)d_in[12];
  float* out = (float*)d_out;

  hipLaunchKernelGGL(rnn_fused_kernel, dim3(BB / 16), dim3(32), 0, stream,
                     x, Wih1, Whh1, bih1, bhh1, Wih2, Whh2, bih2, bhh2,
                     Wfc1, bfc1, Wfc2, bfc2, out);
}
